// TrajectoryAttention_51230369906759
// MI455X (gfx1250) — compile-verified, matrix operand order checked by reading, not run on hardware
//
#include <hip/hip_runtime.h>
#include <hip/hip_bf16.h>

// ---------------- problem constants ----------------
constexpr int cB  = 2;
constexpr int cS  = 1568;
constexpr int cD  = 768;
constexpr int cH  = 12;
constexpr int cF  = 8;
constexpr int cHD = 64;          // cD / cH
constexpr int cN  = 196;         // spatial tokens per frame = S / F
constexpr int cNP = 224;         // padded keys (7 * 32)
constexpr int cBS = cB * cS;     // 3136
constexpr float cSCALE = 0.125f; // HD^-0.5

typedef __attribute__((ext_vector_type(16))) __bf16 v16bf;
typedef __attribute__((ext_vector_type(8)))  float  v8f;

__device__ __forceinline__ v8f wmma_bf16(v16bf a, v16bf b, v8f c) {
  return __builtin_amdgcn_wmma_f32_16x16x32_bf16(
      /*neg_a=*/false, a, /*neg_b=*/false, b,
      /*c_mod=*/(short)0, c, /*reuse_a=*/false, /*reuse_b=*/false);
}

// ============================================================
// Generic GEMM:  C(M,N) = A(M,K) @ W(K,N) + bias
// Block tile 64x64, 4 waves (wave32) each computing 32x32 via
// 2x2 v_wmma_f32_16x16x32_bf16. M%64==0, N%64==0, K%32==0.
// ============================================================
template <bool A_BF16, bool OUT_BF16>
__global__ __launch_bounds__(128) void gemm_wmma_kernel(
    const void* __restrict__ Av, const float* __restrict__ W,
    const float* __restrict__ bias, void* __restrict__ Cv,
    int M, int N, int K) {
  __shared__ __bf16 lA[64 * 32];   // [m][k]
  __shared__ __bf16 lB[64 * 32];   // [n][k]  (W transposed on the fly)

  const int tid  = threadIdx.x;
  const int wave = tid >> 5;
  const int lane = tid & 31;
  const int wm   = (wave >> 1) * 32;
  const int wn   = (wave & 1) * 32;
  const int bm   = blockIdx.y * 64;
  const int bn   = blockIdx.x * 64;
  const int lm   = lane & 15;
  const int hi   = lane >> 4;      // which half-wave
  const int koff = hi * 8;

  v8f acc[2][2] = {};

  for (int k0 = 0; k0 < K; k0 += 32) {
    __syncthreads();
    // ---- stage A tile: 64 rows x 32 k, 16 elems / thread ----
    {
      const int r  = tid >> 1;
      const int c0 = (tid & 1) * 16;
      if constexpr (A_BF16) {
        const __bf16* src = (const __bf16*)Av + (size_t)(bm + r) * K + k0 + c0;
        if (k0 + 32 < K) __builtin_prefetch((const void*)(src + 32), 0, 1);
#pragma unroll
        for (int e = 0; e < 16; ++e) lA[r * 32 + c0 + e] = src[e];
      } else {
        const float* src = (const float*)Av + (size_t)(bm + r) * K + k0 + c0;
        if (k0 + 32 < K) __builtin_prefetch((const void*)(src + 32), 0, 1);
#pragma unroll
        for (int e = 0; e < 16; ++e) lA[r * 32 + c0 + e] = (__bf16)src[e];
      }
    }
    // ---- stage W tile transposed: lB[n][k], 16 elems / thread ----
    {
      const int kk = tid & 31;
      const int n0 = (tid >> 5) * 16;
      const float* src = W + (size_t)(k0 + kk) * N + bn + n0;
#pragma unroll
      for (int e = 0; e < 16; ++e) lB[(n0 + e) * 32 + kk] = (__bf16)src[e];
    }
    __syncthreads();

    // ---- fragments (two contiguous 8x bf16 runs each) ----
    v16bf af[2], bf[2];
#pragma unroll
    for (int i = 0; i < 2; ++i) {
      const __bf16* p = &lA[(wm + i * 16 + lm) * 32 + koff];
#pragma unroll
      for (int e = 0; e < 8; ++e) { af[i][e] = p[e]; af[i][e + 8] = p[e + 16]; }
    }
#pragma unroll
    for (int j = 0; j < 2; ++j) {
      const __bf16* p = &lB[(wn + j * 16 + lm) * 32 + koff];
#pragma unroll
      for (int e = 0; e < 8; ++e) { bf[j][e] = p[e]; bf[j][e + 8] = p[e + 16]; }
    }
#pragma unroll
    for (int i = 0; i < 2; ++i)
#pragma unroll
      for (int j = 0; j < 2; ++j)
        acc[i][j] = wmma_bf16(af[i], bf[j], acc[i][j]);
  }

  // ---- store (C layout: row = r + 8*hi, col = lm) ----
#pragma unroll
  for (int i = 0; i < 2; ++i) {
#pragma unroll
    for (int j = 0; j < 2; ++j) {
      const int n = bn + wn + j * 16 + lm;
      const float bvv = bias ? bias[n] : 0.0f;
#pragma unroll
      for (int r = 0; r < 8; ++r) {
        const int m  = bm + wm + i * 16 + hi * 8 + r;
        const float v = acc[i][j][r] + bvv;
        if constexpr (OUT_BF16)
          ((__bf16*)Cv)[(size_t)m * N + n] = (__bf16)v;
        else
          ((float*)Cv)[(size_t)m * N + n] = v;
      }
    }
  }
}

// ============================================================
// Stage 1: space attention within each frame.
// One block = 2 waves x 16 queries for a single (b, h, f).
// scores = Q K^T (WMMA), masked softmax over 196 keys (padded
// to 224), write space_attn fp32, then P @ V_f (WMMA) -> x bf16.
// x layout: (B, S, F, D) with head h at column h*HD.
// ============================================================
__global__ __launch_bounds__(64) void stage1_attn_kernel(
    const __bf16* __restrict__ qb, const __bf16* __restrict__ kb,
    const __bf16* __restrict__ vb, __bf16* __restrict__ xb,
    float* __restrict__ sattn) {
  const int qt = blockIdx.x;        // 0..48   (32 queries each)
  const int f  = blockIdx.y;        // 0..7
  const int bh = blockIdx.z;        // 0..23
  const int b  = bh / cH;
  const int h  = bh % cH;

  __shared__ __bf16 Vt[cHD * cNP];       // V_f transposed: [d][key]
  __shared__ __bf16 Pl[2][16 * cNP];     // probs per wave: [m][key]

  const int tid  = threadIdx.x;
  const int wave = tid >> 5;
  const int lane = tid & 31;
  const int lm   = lane & 15;
  const int hi   = lane >> 4;
  const int koff = hi * 8;

  // ---- cooperatively stage V_f transposed (zero-padded keys) ----
  {
    const int d = tid;               // 64 threads == 64 head dims
    const __bf16* src = vb + ((size_t)b * cS + (size_t)f * cN) * cD + h * cHD + d;
    for (int key = 0; key < cNP; ++key)
      Vt[d * cNP + key] = (key < cN) ? src[(size_t)key * cD] : (__bf16)0.0f;
  }
  __syncthreads();

  const int q0 = qt * 32 + wave * 16;   // this wave's query base (S=49*32)

  // ---- Q fragments: A 16x64 -> 2 K-steps, contiguous global loads ----
  v16bf qf[2];
#pragma unroll
  for (int ks = 0; ks < 2; ++ks) {
    const __bf16* p = qb + ((size_t)b * cS + q0 + lm) * cD + h * cHD + ks * 32 + koff;
#pragma unroll
    for (int e = 0; e < 8; ++e) { qf[ks][e] = p[e]; qf[ks][e + 8] = p[e + 16]; }
  }

  // ---- scores: 14 key-tiles of 16 ----
  v8f sc[14] = {};
#pragma unroll
  for (int t = 0; t < 14; ++t) {
    const int kin   = t * 16 + lm;        // key within frame for this lane
    const bool valid = kin < cN;
    const __bf16* prow = kb + ((size_t)b * cS + f * cN + kin) * cD + h * cHD;
#pragma unroll
    for (int ks = 0; ks < 2; ++ks) {
      v16bf kf;
      const __bf16* p = prow + ks * 32 + koff;
#pragma unroll
      for (int e = 0; e < 8; ++e) {
        kf[e]     = valid ? p[e]      : (__bf16)0.0f;
        kf[e + 8] = valid ? p[e + 16] : (__bf16)0.0f;
      }
      sc[t] = wmma_bf16(qf[ks], kf, sc[t]);
    }
  }

  // ---- masked softmax per query row (row = r + 8*hi, half-wave reduce) ----
#pragma unroll
  for (int r = 0; r < 8; ++r) {
    float mx = -1e30f;
#pragma unroll
    for (int t = 0; t < 14; ++t) {
      const bool valid = (t * 16 + lm) < cN;
      const float sv = valid ? cSCALE * sc[t][r] : -1e30f;
      sc[t][r] = sv;
      mx = fmaxf(mx, sv);
    }
#pragma unroll
    for (int off = 1; off < 16; off <<= 1) mx = fmaxf(mx, __shfl_xor(mx, off, 32));
    float sum = 0.0f;
#pragma unroll
    for (int t = 0; t < 14; ++t) {
      const float p = __expf(sc[t][r] - mx);
      sc[t][r] = p;
      sum += p;
    }
#pragma unroll
    for (int off = 1; off < 16; off <<= 1) sum += __shfl_xor(sum, off, 32);
    const float inv = 1.0f / sum;
#pragma unroll
    for (int t = 0; t < 14; ++t) sc[t][r] *= inv;
  }

  // ---- emit space_attn (fp32) and stage P (bf16) into LDS ----
  __bf16* Prow = &Pl[wave][0];
#pragma unroll
  for (int t = 0; t < 14; ++t) {
    const int key = t * 16 + lm;
#pragma unroll
    for (int r = 0; r < 8; ++r) {
      const int m   = r + hi * 8;
      const float p = sc[t][r];
      Prow[m * cNP + key] = (__bf16)p;
      if (key < cN)
        sattn[(((size_t)bh * cS + q0 + m) * cF + f) * cN + key] = p;
    }
  }

  // ---- x = P @ V_f : out 16x64, K = 224 = 7 steps ----
  v8f xacc[4] = {};
#pragma unroll
  for (int ks = 0; ks < 7; ++ks) {
    v16bf pf;
    {
      const __bf16* p = &Prow[lm * cNP + ks * 32 + koff];
#pragma unroll
      for (int e = 0; e < 8; ++e) { pf[e] = p[e]; pf[e + 8] = p[e + 16]; }
    }
#pragma unroll
    for (int nt = 0; nt < 4; ++nt) {
      v16bf vf;
      const __bf16* p = &Vt[(nt * 16 + lm) * cNP + ks * 32 + koff];
#pragma unroll
      for (int e = 0; e < 8; ++e) { vf[e] = p[e]; vf[e + 8] = p[e + 16]; }
      xacc[nt] = wmma_bf16(pf, vf, xacc[nt]);
    }
  }

  // ---- store x bf16: xb[((b*S + q)*F + f)*D + h*HD + d] ----
#pragma unroll
  for (int nt = 0; nt < 4; ++nt) {
    const int d = nt * 16 + lm;
#pragma unroll
    for (int r = 0; r < 8; ++r) {
      const int m = r + hi * 8;
      xb[(((size_t)b * cS + q0 + m) * cF + f) * cD + h * cHD + d] = (__bf16)xacc[nt][r];
    }
  }
}

// ============================================================
// Diagonal gather: x_diag[b, s, :] = x[b, s, s/n, :]
// ============================================================
__global__ void diag_gather_kernel(const __bf16* __restrict__ xb,
                                   __bf16* __restrict__ xd) {
  const size_t i = (size_t)blockIdx.x * blockDim.x + threadIdx.x;
  if (i >= (size_t)cBS * cD) return;
  const int d = (int)(i % cD);
  const size_t sb = i / cD;
  const int s = (int)(sb % cS);
  const int b = (int)(sb / cS);
  const int f = s / cN;
  xd[i] = xb[(((size_t)b * cS + s) * cF + f) * cD + d];
}

// ============================================================
// Stage 2: trajectory attention over F=8 frames (tiny, VALU).
// One thread handles one (b, s, h). Writes bf16 pre-projection.
// ============================================================
__global__ __launch_bounds__(128) void stage2_attn_kernel(
    const float* __restrict__ q2, const float* __restrict__ kv,
    __bf16* __restrict__ ao) {
  const int i = blockIdx.x * blockDim.x + threadIdx.x;
  if (i >= cB * cS * cH) return;
  const int h = i % cH;
  const int sb = i / cH;
  const int s = sb % cS;
  const int b = sb / cS;

  const float* qv = q2 + ((size_t)b * cS + s) * cD + h * cHD;
  const size_t rowkv = ((size_t)b * cS + s) * cF;

  float logit[cF];
#pragma unroll
  for (int f = 0; f < cF; ++f) {
    const float* kf = kv + (rowkv + f) * (2 * cD) + h * cHD;
    float acc = 0.0f;
    for (int d = 0; d < cHD; ++d) acc += qv[d] * kf[d];
    logit[f] = acc * cSCALE;    // scale folded from q2 * HD^-0.5
  }
  float mx = logit[0];
#pragma unroll
  for (int f = 1; f < cF; ++f) mx = fmaxf(mx, logit[f]);
  float sum = 0.0f;
#pragma unroll
  for (int f = 0; f < cF; ++f) { logit[f] = __expf(logit[f] - mx); sum += logit[f]; }
  const float inv = 1.0f / sum;

  __bf16* dst = ao + ((size_t)b * cS + s) * cD + h * cHD;
  for (int d = 0; d < cHD; ++d) {
    float acc = 0.0f;
#pragma unroll
    for (int f = 0; f < cF; ++f)
      acc += logit[f] * kv[(rowkv + f) * (2 * cD) + cD + h * cHD + d];
    dst[d] = (__bf16)(acc * inv);
  }
}

// ============================================================
// Host-side orchestration
// ============================================================
extern "C" void kernel_launch(void* const* d_in, const int* in_sizes, int n_in,
                              void* d_out, int out_size, void* d_ws, size_t ws_size,
                              hipStream_t stream) {
  const float* query = (const float*)d_in[0];
  const float* key_t = (const float*)d_in[1];
  const float* value = (const float*)d_in[2];
  const float* Wq    = (const float*)d_in[3];
  const float* bq    = (const float*)d_in[4];
  const float* Wk    = (const float*)d_in[5];
  const float* bk    = (const float*)d_in[6];
  const float* Wv    = (const float*)d_in[7];
  const float* bv    = (const float*)d_in[8];
  const float* Wpq   = (const float*)d_in[9];
  const float* bpq   = (const float*)d_in[10];
  const float* Wpkv  = (const float*)d_in[11];
  const float* bpkv  = (const float*)d_in[12];
  const float* Wp    = (const float*)d_in[13];
  const float* bp    = (const float*)d_in[14];
  (void)in_sizes; (void)n_in; (void)out_size; (void)ws_size;

  float* out   = (float*)d_out;                       // (B, S, D) fp32
  float* sattn = out + (size_t)cBS * cD;              // (B*H, S, F, n) fp32

  // ---- carve workspace ----
  char* ws = (char*)d_ws;
  size_t off = 0;
  auto carve = [&](size_t bytes) -> void* {
    void* p = ws + off;
    off += (bytes + 255) & ~(size_t)255;
    return p;
  };
  __bf16* qb   = (__bf16*)carve((size_t)cBS * cD * 2);
  __bf16* kb   = (__bf16*)carve((size_t)cBS * cD * 2);
  __bf16* vb   = (__bf16*)carve((size_t)cBS * cD * 2);
  __bf16* xbuf = (__bf16*)carve((size_t)cBS * cF * cD * 2);   // (B,S,F,D)
  __bf16* xdg  = (__bf16*)carve((size_t)cBS * cD * 2);        // x_diag
  __bf16* ao   = (__bf16*)carve((size_t)cBS * cD * 2);        // stage2 out
  float*  q2f  = (float*)carve((size_t)cBS * cD * 4);
  float*  kvf  = (float*)carve((size_t)cBS * cF * 2 * cD * 4);

  // ---- q/k/v projections: fp32 in -> bf16 out ----
  dim3 gProj(cD / 64, cBS / 64);
  gemm_wmma_kernel<false, true><<<gProj, 128, 0, stream>>>(query, Wq, bq, qb, cBS, cD, cD);
  gemm_wmma_kernel<false, true><<<gProj, 128, 0, stream>>>(key_t, Wk, bk, kb, cBS, cD, cD);
  gemm_wmma_kernel<false, true><<<gProj, 128, 0, stream>>>(value, Wv, bv, vb, cBS, cD, cD);

  // ---- stage 1 space attention ----
  stage1_attn_kernel<<<dim3(cS / 32, cF, cB * cH), 64, 0, stream>>>(qb, kb, vb, xbuf, sattn);

  // ---- diagonal gather ----
  {
    const size_t tot = (size_t)cBS * cD;
    diag_gather_kernel<<<(unsigned)((tot + 255) / 256), 256, 0, stream>>>(xbuf, xdg);
  }

  // ---- q2 = x_diag @ Wpq + bpq (bf16 in -> fp32 out) ----
  gemm_wmma_kernel<true, false><<<gProj, 128, 0, stream>>>(xdg, Wpq, bpq, q2f, cBS, cD, cD);

  // ---- kv = x @ Wpkv + bpkv : M = B*S*F, N = 2D ----
  gemm_wmma_kernel<true, false><<<dim3(2 * cD / 64, cBS * cF / 64), 128, 0, stream>>>(
      xbuf, Wpkv, bpkv, kvf, cBS * cF, 2 * cD, cD);

  // ---- stage 2 trajectory attention ----
  {
    const int nt = cB * cS * cH;
    stage2_attn_kernel<<<(nt + 127) / 128, 128, 0, stream>>>(q2f, kvf, ao);
  }

  // ---- final projection: out = ao @ Wp + bp (fp32 to d_out) ----
  gemm_wmma_kernel<true, false><<<gProj, 128, 0, stream>>>(ao, Wp, bp, out, cBS, cD, cD);
}
